// VectorQuantizer_6399501271151
// MI455X (gfx1250) — matrix operand order repaired and checked by reading, not compiled, not run on hardware
//
#include <hip/hip_runtime.h>
#include <hip/hip_bf16.h>

// ---------------- Types for CDNA5 WMMA ----------------
typedef __attribute__((ext_vector_type(16))) __bf16 v16bf;
typedef __attribute__((ext_vector_type(8)))  float  v8f;

union Frag16 {            // 16 x bf16 = 32 bytes = 8 VGPRs
    unsigned u[8];
    uint4    u4[2];
    v16bf    v;
};

// ---------------- Problem constants ----------------
#define VQ_D        256
#define VQ_N        8192
#define VQ_T        32768           // 8*4096 tokens
#define KSTEPS      8               // 256 / 32 (K per bf16 WMMA)
#define NCHUNK      64              // codes staged in LDS per iteration
#define NSPLIT      4               // N-dimension split across blocks
#define NPART       (VQ_N / NSPLIT) // 2048 codes per block
#define ZQ_ELEMS    (VQ_T * VQ_D)   // 8388608
#define LOSS_OFF    ZQ_ELEMS
#define IDX_OFF     (ZQ_ELEMS + 1)

// ---------------- Helpers ----------------
__device__ __forceinline__ unsigned short f2bf(float f) {
    unsigned u = __float_as_uint(f);
    unsigned r = 0x7FFFu + ((u >> 16) & 1u);   // round-to-nearest-even
    return (unsigned short)((u + r) >> 16);
}
__device__ __forceinline__ unsigned pack2bf(float a, float b) {
    return (unsigned)f2bf(a) | ((unsigned)f2bf(b) << 16);
}
// monotonic float -> unsigned mapping (preserves <)
__device__ __forceinline__ unsigned orderF32(float f) {
    unsigned u = __float_as_uint(f);
    return (u & 0x80000000u) ? ~u : (u ^ 0x80000000u);
}

// ---------------- Kernel 1: emb -> bf16, ||e||^2, init keys ----------------
__global__ __launch_bounds__(256)
void vq_prep(const float* __restrict__ emb,
             unsigned short* __restrict__ emb_bf,
             float* __restrict__ enorm,
             unsigned long long* __restrict__ keys) {
    const int n = blockIdx.x;          // code row
    const int d = threadIdx.x;         // dim
    float v = emb[(size_t)n * VQ_D + d];
    emb_bf[(size_t)n * VQ_D + d] = f2bf(v);
    float s = v * v;
    #pragma unroll
    for (int off = 16; off >= 1; off >>= 1) s += __shfl_down(s, off, 32);
    __shared__ float red[8];
    if ((d & 31) == 0) red[d >> 5] = s;
    __syncthreads();
    if (d == 0) {
        float t = 0.f;
        #pragma unroll
        for (int i = 0; i < 8; ++i) t += red[i];
        enorm[n] = t;
    }
    if (blockIdx.x < (VQ_T / 256))     // first 128 blocks init the 32768 keys
        keys[blockIdx.x * 256 + threadIdx.x] = 0xFFFFFFFFFFFFFFFFull;
}

// ---------------- Kernel 2: WMMA distance GEMM + argmin ----------------
// grid = 128 token-blocks * NSPLIT,  block = 256 threads (8 waves)
// wave handles 32 tokens (2 A tiles of 16), block handles NPART codes.
__global__ __launch_bounds__(256)
void vq_argmin(const float* __restrict__ z,
               const unsigned short* __restrict__ emb_bf,
               const float* __restrict__ enorm,
               unsigned long long* __restrict__ keys) {
    __shared__ __align__(16) unsigned lds_e[NCHUNK * (VQ_D / 2)]; // 64x128 dw = 32 KB

    const int tid   = threadIdx.x;
    const int lane  = tid & 31;
    const int wave  = tid >> 5;
    const int tblk  = blockIdx.x / NSPLIT;
    const int nsp   = blockIdx.x % NSPLIT;
    const int tbase = tblk * 256 + wave * 32;
    const int col   = lane & 15;
    const int khalf = lane >> 4;

    // ---- Build A fragments (bf16) in registers, full K=256, 2 token tiles ----
    Frag16 a0[KSTEPS], a1[KSTEPS];
    {
        const float2* zr0 = reinterpret_cast<const float2*>(z + (size_t)(tbase + col)      * VQ_D);
        const float2* zr1 = reinterpret_cast<const float2*>(z + (size_t)(tbase + 16 + col) * VQ_D);
        #pragma unroll
        for (int ks = 0; ks < KSTEPS; ++ks) {
            #pragma unroll
            for (int i = 0; i < 8; ++i) {
                // ISA 16-bit A layout: lanes 0-15 K 0..7 / 16..23, lanes 16-31 K 8..15 / 24..31
                const int koff = (i < 4) ? (khalf * 8 + 2 * i)
                                         : (16 + khalf * 8 + 2 * (i - 4));
                const int kp = (ks * 32 + koff) >> 1;   // float2 index
                float2 p0 = zr0[kp];
                float2 p1 = zr1[kp];
                a0[ks].u[i] = pack2bf(p0.x, p0.y);
                a1[ks].u[i] = pack2bf(p1.x, p1.y);
            }
        }
    }

    float    mind[2][8];
    unsigned minn[2][8];
    #pragma unroll
    for (int mt = 0; mt < 2; ++mt)
        #pragma unroll
        for (int r = 0; r < 8; ++r) { mind[mt][r] = 3.4e38f; minn[mt][r] = 0u; }

    const unsigned* ebase = reinterpret_cast<const unsigned*>(emb_bf);

    for (int chunk = 0; chunk < NPART / NCHUNK; ++chunk) {       // 32 iters
        const int n0 = nsp * NPART + chunk * NCHUNK;
        // stage NCHUNK code rows (bf16) into LDS: 8192 dwords = 2048 uint4
        {
            const uint4* src = reinterpret_cast<const uint4*>(ebase + (size_t)n0 * (VQ_D / 2));
            uint4*       dst = reinterpret_cast<uint4*>(lds_e);
            #pragma unroll
            for (int i = tid; i < NCHUNK * (VQ_D / 8); i += 256) dst[i] = src[i];
        }
        __syncthreads();

        #pragma unroll
        for (int tile = 0; tile < NCHUNK / 16; ++tile) {          // 4 x 16 codes
            v8f c0 = {}; v8f c1 = {};
            const int rowbase = tile * 16 + col;                  // code row in LDS
            const uint4* lrow = reinterpret_cast<const uint4*>(lds_e) + rowbase * (VQ_D / 8);
            #pragma unroll
            for (int ks = 0; ks < KSTEPS; ++ks) {
                // B layout matching A: lanes 0-15 K 0..7 / 16..23, lanes 16-31 K 8..15 / 24..31; N = lane&15
                Frag16 b;
                const uint4* lp = lrow + ks * 4 + khalf;
                b.u4[0] = lp[0];
                b.u4[1] = lp[2];
                c0 = __builtin_amdgcn_wmma_f32_16x16x32_bf16(
                        false, a0[ks].v, false, b.v, (short)0, c0, false, false);
                c1 = __builtin_amdgcn_wmma_f32_16x16x32_bf16(
                        false, a1[ks].v, false, b.v, (short)0, c1, false, false);
            }
            const int   n  = n0 + tile * 16 + col;
            const float en = enorm[n];
            #pragma unroll
            for (int r = 0; r < 8; ++r) {
                float d0 = en - 2.0f * c0[r];   // ||e||^2 - 2 z.e  (||z||^2 const per row)
                if (d0 < mind[0][r]) { mind[0][r] = d0; minn[0][r] = (unsigned)n; }
                float d1 = en - 2.0f * c1[r];
                if (d1 < mind[1][r]) { mind[1][r] = d1; minn[1][r] = (unsigned)n; }
            }
        }
        __syncthreads();
    }

    // ---- cross-lane argmin within 16-lane halves, then global merge ----
    #pragma unroll
    for (int mt = 0; mt < 2; ++mt) {
        #pragma unroll
        for (int r = 0; r < 8; ++r) {
            unsigned long long key =
                ((unsigned long long)orderF32(mind[mt][r]) << 32) | minn[mt][r];
            #pragma unroll
            for (int off = 8; off >= 1; off >>= 1) {
                unsigned lo = (unsigned)key, hi = (unsigned)(key >> 32);
                unsigned olo = __shfl_xor(lo, off, 32);
                unsigned ohi = __shfl_xor(hi, off, 32);
                unsigned long long ok = ((unsigned long long)ohi << 32) | olo;
                if (ok < key) key = ok;
            }
            if (col == 0) {
                // C layout: VGPR r -> M = r + 8*(lane>>4)
                const int token = tbase + mt * 16 + r + khalf * 8;
                atomicMin(&keys[token], key);
            }
        }
    }
}

// ---------------- Kernel 3: fp32 gather, z_q output, loss partials ----------------
__global__ __launch_bounds__(256)
void vq_gather(const float* __restrict__ z,
               const float* __restrict__ emb,
               const unsigned long long* __restrict__ keys,
               float* __restrict__ out,
               float* __restrict__ partials) {
    const int t = blockIdx.x;
    const int d = threadIdx.x;
    const unsigned idx = (unsigned)(keys[t] & 0xFFFFFFFFull);
    const float e  = emb[(size_t)idx * VQ_D + d];
    const float zv = z[(size_t)t * VQ_D + d];
    out[(size_t)t * VQ_D + d] = e;          // z + sg(z_q - z) == z_q numerically
    float df = e - zv;
    float s = df * df;
    #pragma unroll
    for (int off = 16; off >= 1; off >>= 1) s += __shfl_down(s, off, 32);
    __shared__ float red[8];
    if ((d & 31) == 0) red[d >> 5] = s;
    __syncthreads();
    if (d == 0) {
        float tt = 0.f;
        #pragma unroll
        for (int i = 0; i < 8; ++i) tt += red[i];
        partials[t] = tt;
        out[IDX_OFF + t] = (float)idx;      // encoding indices
    }
}

// ---------------- Kernel 4: deterministic loss reduction ----------------
__global__ __launch_bounds__(256)
void vq_finalize(const float* __restrict__ partials, float* __restrict__ out) {
    __shared__ float red[256];
    float s = 0.f;
    for (int i = threadIdx.x; i < VQ_T; i += 256) s += partials[i];  // fixed order
    red[threadIdx.x] = s;
    __syncthreads();
    #pragma unroll
    for (int stride = 128; stride >= 1; stride >>= 1) {
        if (threadIdx.x < stride) red[threadIdx.x] += red[threadIdx.x + stride];
        __syncthreads();
    }
    if (threadIdx.x == 0)
        out[LOSS_OFF] = 1.25f * red[0] / (float)ZQ_ELEMS;  // (1+BETA)*mean
}

// ---------------- Launch ----------------
extern "C" void kernel_launch(void* const* d_in, const int* in_sizes, int n_in,
                              void* d_out, int out_size, void* d_ws, size_t ws_size,
                              hipStream_t stream) {
    (void)in_sizes; (void)n_in; (void)out_size; (void)ws_size;
    const float* z   = (const float*)d_in[0];   // [8,4096,256] f32
    const float* emb = (const float*)d_in[1];   // [8192,256]  f32
    float* out = (float*)d_out;

    char* ws = (char*)d_ws;
    unsigned short*     emb_bf   = (unsigned short*)ws;                       // 4 MB
    float*              enorm    = (float*)(ws + (4u << 20));                 // 32 KB
    unsigned long long* keys     = (unsigned long long*)(ws + (4u << 20) + (32u << 10));   // 256 KB
    float*              partials = (float*)(ws + (4u << 20) + (32u << 10) + (256u << 10)); // 128 KB

    vq_prep<<<VQ_N, 256, 0, stream>>>(emb, emb_bf, enorm, keys);
    vq_argmin<<<(VQ_T / 256) * NSPLIT, 256, 0, stream>>>(z, emb_bf, enorm, keys);
    vq_gather<<<VQ_T, 256, 0, stream>>>(z, emb, keys, out, partials);
    vq_finalize<<<1, 256, 0, stream>>>(partials, out);
}
